// NeuralScorer_66786741453010
// MI455X (gfx1250) — matrix operand order repaired and checked by reading, not compiled, not run on hardware
//
#include <hip/hip_runtime.h>
#include <hip/hip_bf16.h>

typedef _Float16 v8h  __attribute__((ext_vector_type(8)));
typedef _Float16 v16h __attribute__((ext_vector_type(16)));
typedef float    v8f  __attribute__((ext_vector_type(8)));

#define B_    8
#define TG_   4096
#define M_    3
#define E_    256
#define F_    768          // M*E
#define H_    384          // hidden
#define KT_   24           // 768 / 32 (K per WMMA)
#define NT_   24           // 384 / 16 (N per WMMA)
#define KOUT_ 512
#define XPAD_ 776          // 768 + 8 f16 pad -> row stride 1552B, breaks bank aliasing
#define NEGB_ 1.0e9f

// ---------------------------------------------------------------------------
// Repack W1 [768,384] f32 row-major -> f16 B-fragment layout for
// v_wmma_f32_16x16x32_f16: per (kt,nt) tile, lane L holds column N=L%16,
// K = kt*32 + (L/16)*16 + j for j=0..15 (16 contiguous f16 per lane).
// ---------------------------------------------------------------------------
__global__ void prep_w1_kernel(const float* __restrict__ W1,
                               _Float16* __restrict__ out) {
    int tid  = blockIdx.x * blockDim.x + threadIdx.x;   // 294912 total
    int j    = tid & 15;
    int g    = tid >> 4;
    int lane = g & 31;
    int pair = g >> 5;            // kt*NT_ + nt
    int nt   = pair % NT_;
    int kt   = pair / NT_;
    int f    = kt * 32 + (lane >> 4) * 16 + j;
    int col  = nt * 16 + (lane & 15);
    out[tid] = (_Float16)W1[f * H_ + col];
}

// ---------------------------------------------------------------------------
// Fused embed-gather + MLP + mask-bias score kernel.
// 2 waves / block, each wave owns one 16-grounding tile.
// ---------------------------------------------------------------------------
__global__ __launch_bounds__(64) void score_kernel(
    const int*      __restrict__ body,   // [B,TG,M,3]
    const int*      __restrict__ mask,   // [B,TG]
    const float*    __restrict__ subj,   // [V,E]
    const float*    __restrict__ obj,    // [V,E]
    const _Float16* __restrict__ w1f,    // repacked fragments
    const float*    __restrict__ b1,     // [H]
    const float*    __restrict__ W2,     // [H,1]
    const float*    __restrict__ b2,     // [1]
    float*          __restrict__ scores) // [B,TG]
{
    __shared__ __align__(16) _Float16 Xs[2][16][XPAD_];

    const int w    = threadIdx.x >> 5;
    const int lane = threadIdx.x & 31;
    const int tile = blockIdx.x * 2 + w;     // 2048 tiles total
    const int b    = tile >> 8;              // 256 tiles per batch
    const int t0   = (tile & 255) << 4;

    // ---- stage X tile: (subj[s] + obj[o]) * active, f32 -> f16 into LDS ----
    for (int it = 0; it < 48; ++it) {        // 16 rows * 3 atoms
        int row = it / 3, m = it % 3;
        const int* bp = body + (((size_t)(b * TG_ + (t0 + row)) * M_ + m) * 3);
        int   a0  = bp[0], si = bp[1], oi = bp[2];
        float act = (a0 != 0) ? 1.0f : 0.0f;
        const float4* sp = (const float4*)(subj + (size_t)si * E_) + lane * 2;
        const float4* op = (const float4*)(obj  + (size_t)oi * E_) + lane * 2;
        float4 s0 = sp[0], s1 = sp[1];
        float4 o0 = op[0], o1 = op[1];
        v8h hv;
        hv[0] = (_Float16)((s0.x + o0.x) * act);
        hv[1] = (_Float16)((s0.y + o0.y) * act);
        hv[2] = (_Float16)((s0.z + o0.z) * act);
        hv[3] = (_Float16)((s0.w + o0.w) * act);
        hv[4] = (_Float16)((s1.x + o1.x) * act);
        hv[5] = (_Float16)((s1.y + o1.y) * act);
        hv[6] = (_Float16)((s1.z + o1.z) * act);
        hv[7] = (_Float16)((s1.w + o1.w) * act);
        *(v8h*)&Xs[w][row][m * E_ + lane * 8] = hv;
    }
    __syncthreads();

    // ---- WMMA main loop: h = relu(X@W1 + b1); s = h @ W2 (fused) ----------
    const int half = lane >> 4;     // A-fragment half (K groups)
    const int rrow = lane & 15;     // A row / C column index
    const float b2v = b2[0];
    float sacc[8];
#pragma unroll
    for (int r = 0; r < 8; ++r) sacc[r] = 0.0f;

    for (int nt = 0; nt < NT_; ++nt) {
        v8f acc = {};
        for (int kt = 0; kt < KT_; ++kt) {
            int kb = kt * 32 + half * 16;
            v8h alo = *(const v8h*)&Xs[w][rrow][kb];
            v8h ahi = *(const v8h*)&Xs[w][rrow][kb + 8];
            v16h a = __builtin_shufflevector(alo, ahi,
                        0,1,2,3,4,5,6,7,8,9,10,11,12,13,14,15);
            v16h bf = *(const v16h*)(w1f +
                        ((size_t)(kt * NT_ + nt) * 32 + lane) * 16);
            acc = __builtin_amdgcn_wmma_f32_16x16x32_f16(
                      false, a, false, bf, (short)0, acc, false, false);
        }
        float bb  = b1[nt * 16 + rrow];
        float w2v = W2[nt * 16 + rrow];
#pragma unroll
        for (int r = 0; r < 8; ++r) {
            float h = acc[r] + bb;
            h = h > 0.0f ? h : 0.0f;
            sacc[r] += h * w2v;
        }
    }

    // reduce over the 16 lanes sharing a C-matrix half (wave32 butterfly)
#pragma unroll
    for (int r = 0; r < 8; ++r) {
        sacc[r] += __shfl_xor(sacc[r], 1, 32);
        sacc[r] += __shfl_xor(sacc[r], 2, 32);
        sacc[r] += __shfl_xor(sacc[r], 4, 32);
        sacc[r] += __shfl_xor(sacc[r], 8, 32);
    }
    if ((lane & 15) == 0) {
#pragma unroll
        for (int r = 0; r < 8; ++r) {
            int t = t0 + r + 8 * half;       // half0 -> rows 0..7, half1 -> 8..15
            float sc = sacc[r] + b2v +
                       ((float)mask[b * TG_ + t] - 1.0f) * NEGB_;
            scores[b * TG_ + t] = sc;
        }
    }
}

// ---------------------------------------------------------------------------
// Per-batch bitonic sort (descending, low-index tie-break) of 4096 keys.
// ---------------------------------------------------------------------------
__global__ __launch_bounds__(1024) void topk_sort_kernel(
    const float* __restrict__ scores, int* __restrict__ sidx)
{
    __shared__ unsigned long long keys[TG_];
    const int b   = blockIdx.x;
    const int tid = threadIdx.x;

    for (int i = tid; i < TG_; i += 1024) {
        unsigned int bits = __float_as_uint(scores[b * TG_ + i]);
        unsigned int u = (bits & 0x80000000u) ? ~bits : (bits | 0x80000000u);
        keys[i] = ((unsigned long long)u << 32) |
                  (unsigned int)(0xFFFFFFFFu - (unsigned int)i);
    }
    __syncthreads();

    for (int k = 2; k <= TG_; k <<= 1) {
        for (int j = k >> 1; j > 0; j >>= 1) {
            for (int i = tid; i < TG_; i += 1024) {
                int ixj = i ^ j;
                if (ixj > i) {
                    unsigned long long a = keys[i], c = keys[ixj];
                    bool desc = ((i & k) == 0);
                    if (desc ? (a < c) : (a > c)) { keys[i] = c; keys[ixj] = a; }
                }
            }
            __syncthreads();
        }
    }
    for (int i = tid; i < KOUT_; i += 1024)
        sidx[b * KOUT_ + i] =
            (int)(0xFFFFFFFFu - (unsigned int)(keys[i] & 0xFFFFFFFFu));
}

// ---------------------------------------------------------------------------
// Gather outputs: body_sel | mask_sel | rule_sel | top_scores (flat concat).
// ---------------------------------------------------------------------------
__global__ void gather_out_kernel(
    const int* __restrict__ body, const int* __restrict__ mask,
    const int* __restrict__ rule, const float* __restrict__ scores,
    const int* __restrict__ sidx,
    int* __restrict__ outi, float* __restrict__ outf)
{
    int tid = blockIdx.x * blockDim.x + threadIdx.x;   // 4096 = B*KOUT
    int b = tid >> 9, k = tid & 511;
    int idx = sidx[b * KOUT_ + k];

    const int* src = body + (size_t)(b * TG_ + idx) * 9;
    int*       dst = outi + (size_t)(b * KOUT_ + k) * 9;
#pragma unroll
    for (int j = 0; j < 9; ++j) dst[j] = src[j];

    const int OFF = B_ * KOUT_ * 9;                    // 36864
    outi[OFF + b * KOUT_ + k]               = mask[b * TG_ + idx];
    outi[OFF + B_ * KOUT_ + b * KOUT_ + k]  = rule[b * TG_ + idx];
    outf[OFF + 2 * B_ * KOUT_ + b * KOUT_ + k] = scores[b * TG_ + idx];
}

// ---------------------------------------------------------------------------
extern "C" void kernel_launch(void* const* d_in, const int* in_sizes, int n_in,
                              void* d_out, int out_size, void* d_ws, size_t ws_size,
                              hipStream_t stream) {
    const int*   body = (const int*)  d_in[0];
    const int*   mask = (const int*)  d_in[1];
    const int*   rule = (const int*)  d_in[2];
    const float* subj = (const float*)d_in[3];
    const float* obj  = (const float*)d_in[4];
    const float* W1   = (const float*)d_in[5];
    const float* b1   = (const float*)d_in[6];
    const float* W2   = (const float*)d_in[7];
    const float* b2   = (const float*)d_in[8];

    char* wsb = (char*)d_ws;
    _Float16* w1f    = (_Float16*)wsb;                      // 589824 B
    float*    scores = (float*)(wsb + 589824);              // 131072 B
    int*      sidx   = (int*)  (wsb + 589824 + 131072);     // 16384 B

    prep_w1_kernel<<<1152, 256, 0, stream>>>(W1, w1f);      // 294912 threads
    score_kernel<<<1024, 64, 0, stream>>>(body, mask, subj, obj,
                                          w1f, b1, W2, b2, scores);
    topk_sort_kernel<<<B_, 1024, 0, stream>>>(scores, sidx);
    gather_out_kernel<<<16, 256, 0, stream>>>(body, mask, rule, scores, sidx,
                                              (int*)d_out, (float*)d_out);
}
